// PaperRNNHead_59072980189899
// MI455X (gfx1250) — matrix operand order repaired and checked by reading, not compiled, not run on hardware
//
#include <hip/hip_runtime.h>
#include <hip/hip_bf16.h>

typedef __attribute__((ext_vector_type(16))) __bf16 v16bf;
typedef __attribute__((ext_vector_type(8)))  float  v8f;

#define HIDN 128
#define CTXN 512
#define INDIM 514
#define PREDN 12
#define OBSN 8
#define WAVES 8
#define RPW 16                    // rows per wave
#define RPB (WAVES * RPW)         // rows per block = 128

// ---------- bf16 helpers: use native gfx1250 converts ----------
__device__ __forceinline__ __bf16 f2bf(float f) { return (__bf16)f; }
__device__ __forceinline__ float bf2f(__bf16 h) { return (float)h; }

// native V_TANH_F32 (TRANS op on CDNA5) when available; libm fallback otherwise
__device__ __forceinline__ float fast_tanh(float x) {
#if __has_builtin(__builtin_amdgcn_tanhf)
    return __builtin_amdgcn_tanhf(x);
#elif __has_builtin(__builtin_amdgcn_tanh_f32)
    return __builtin_amdgcn_tanh_f32(x);
#else
    return tanhf(x);
#endif
}

// load 8 contiguous bf16 (16B) from LDS into fragment slots [base, base+8)
__device__ __forceinline__ void load8(const __bf16* p, v16bf& frag, int base) {
    union { uint4 u; __bf16 h[8]; } t;
    t.u = *(const uint4*)p;
#pragma unroll
    for (int i = 0; i < 8; ++i) frag[base + i] = t.h[i];
}

// A fragment (16x32 bf16, M=rows): lane m=lane&15, half=lane>>4.
// VGPR v holds K = (v/4)*16 + half*8 + (v%4)*2 (+1)  ->  frag[0..7]=k0..k0+7, frag[8..15]=k0+16..k0+23
__device__ __forceinline__ v16bf loadA(const __bf16* tile /*[16][128]*/, int lane, int kt) {
    int m = lane & 15, half = lane >> 4;
    const __bf16* p = tile + m * HIDN + kt * 32 + half * 8;
    v16bf a;
    load8(p, a, 0);
    load8(p + 16, a, 8);
    return a;
}

// B fragment (32x16 bf16, B = W^T): lane n=lane&15 (+16*nt), half=lane>>4.
// frag[j] = W[n][kt*32 + half*8 + j] (j<8), W[n][kt*32 + 16 + half*8 + j-8] (j>=8) -> two 16B loads
__device__ __forceinline__ v16bf loadB(const __bf16* w, int ldw, int lane, int nt, int kt) {
    int n = (lane & 15) + nt * 16, half = lane >> 4;
    const __bf16* p = w + n * ldw + kt * 32 + half * 8;
    v16bf b;
    load8(p, b, 0);
    load8(p + 16, b, 8);
    return b;
}

// A fragment straight from global f32 (ctx projection phase)
__device__ __forceinline__ v16bf loadA_g(const float* row, int kt, int half) {
    int k0 = kt * 32 + half * 8;
    float4 c0 = *(const float4*)(row + k0);
    float4 c1 = *(const float4*)(row + k0 + 4);
    float4 c2 = *(const float4*)(row + k0 + 16);
    float4 c3 = *(const float4*)(row + k0 + 20);
    v16bf a;
    a[0]=f2bf(c0.x); a[1]=f2bf(c0.y); a[2]=f2bf(c0.z); a[3]=f2bf(c0.w);
    a[4]=f2bf(c1.x); a[5]=f2bf(c1.y); a[6]=f2bf(c1.z); a[7]=f2bf(c1.w);
    a[8]=f2bf(c2.x); a[9]=f2bf(c2.y); a[10]=f2bf(c2.z); a[11]=f2bf(c2.w);
    a[12]=f2bf(c3.x); a[13]=f2bf(c3.y); a[14]=f2bf(c3.z); a[15]=f2bf(c3.w);
    return a;
}

// [16,128] x [128,128]^T tile GEMM: 4 k-steps of v_wmma_f32_16x16x32_bf16
__device__ __forceinline__ v8f gemm_tile(const __bf16* Atile, const __bf16* W,
                                         int lane, int nt, v8f acc) {
#pragma unroll
    for (int kt = 0; kt < 4; ++kt) {
        v16bf a = loadA(Atile, lane, kt);
        v16bf b = loadB(W, HIDN, lane, nt, kt);
        acc = __builtin_amdgcn_wmma_f32_16x16x32_bf16(false, a, false, b,
                                                      (short)0, acc, false, false);
    }
    return acc;
}

// scatter a D tile (f32, lane n=lane&15, VGPR v -> m = v + 8*(lane>>4)) into
// bf16 A-layout LDS tile [m][feature]
__device__ __forceinline__ void storeD(__bf16* tile, int lane, int nt, const v8f& d) {
    int n = (lane & 15) + nt * 16, half = lane >> 4;
#pragma unroll
    for (int v = 0; v < 8; ++v) tile[(v + 8 * half) * HIDN + n] = f2bf(d[v]);
}

extern "C" __global__ void __launch_bounds__(256, 1)
rnn_head_kernel(const float* __restrict__ raw_ctx, const float* __restrict__ obs_traj,
                const float* __restrict__ Wih0, const float* __restrict__ Whh0,
                const float* __restrict__ bih0, const float* __restrict__ bhh0,
                const float* __restrict__ Wih1, const float* __restrict__ Whh1,
                const float* __restrict__ bih1, const float* __restrict__ bhh1,
                const float* __restrict__ Wih2, const float* __restrict__ Whh2,
                const float* __restrict__ bih2, const float* __restrict__ bhh2,
                const float* __restrict__ fcW,  const float* __restrict__ fcb,
                float* __restrict__ out, int Bn)
{
    extern __shared__ __align__(16) char smem[];
    // weight region: 81920 bf16 = 160KB. Used first as Wctx[128][512] (128KB),
    // then re-staged as 5x W[128][128] for the recurrence.
    __bf16* wreg = (__bf16*)smem;
    __bf16* act  = (__bf16*)(smem + 163840);          // 8 waves * 3 tiles * 16*128 bf16 = 96KB
    float*  wpos = (float*)(smem + 262144);           // [128][2]
    float*  b1s  = wpos + 256;                        // [128]
    float*  b2s  = b1s + 128;                         // [128]
    float*  fcwl = b2s + 128;                         // [2][128]
    float*  fcbl = fcwl + 256;                        // [2] (+pad)
    float*  posl = fcbl + 4;                          // [8 waves][16][2]

    const int tid  = threadIdx.x;
    const int lane = tid & 31;
    const int wave = tid >> 5;
    const int half = lane >> 4;
    const int rb   = blockIdx.x * RPB + wave * RPW;
    if (rb >= Bn) return;

    // ---- phase 0: stage Wctx (bf16) + small params ----
    for (int i = tid; i < HIDN * CTXN; i += 256) {
        int n = i >> 9, k = i & 511;
        wreg[i] = f2bf(Wih0[n * INDIM + k]);
    }
    {   // one element per thread (256 each)
        int n = tid >> 1, j = tid & 1;
        wpos[tid] = Wih0[n * INDIM + CTXN + j];
        fcwl[tid] = fcW[tid];
    }
    if (tid < 128) { b1s[tid] = bih1[tid] + bhh1[tid]; b2s[tid] = bih2[tid] + bhh2[tid]; }
    if (tid < 2)   fcbl[tid] = fcb[tid];
    __syncthreads();

    // ---- phase 1: step-invariant context projection, held in VGPRs ----
    const float* xrow = raw_ctx + (size_t)(rb + (lane & 15)) * CTXN;
    v8f cacc[8];
#pragma unroll
    for (int nt = 0; nt < 8; ++nt) {
        int n = (lane & 15) + nt * 16;
        float bias = bih0[n] + bhh0[n];
        v8f acc;
#pragma unroll
        for (int v = 0; v < 8; ++v) acc[v] = bias;
#pragma unroll
        for (int kt = 0; kt < 16; ++kt) {       // K = 512
            v16bf a = loadA_g(xrow, kt, half);
            v16bf b = loadB(wreg, CTXN, lane, nt, kt);
            acc = __builtin_amdgcn_wmma_f32_16x16x32_bf16(false, a, false, b,
                                                          (short)0, acc, false, false);
        }
        cacc[nt] = acc;
    }
    __syncthreads();   // everyone done reading Wctx

    // ---- phase 2: re-stage the 5 recurrent weight matrices (bf16) ----
    {
        const float* srcs[5] = { Whh0, Wih1, Whh1, Wih2, Whh2 };
        for (int mtx = 0; mtx < 5; ++mtx) {
            const float* s = srcs[mtx];
            __bf16* d = wreg + mtx * HIDN * HIDN;
            for (int i = tid; i < HIDN * HIDN; i += 256) d[i] = f2bf(s[i]);
        }
    }
    // per-wave activation tiles; zero-init hidden states
    __bf16* h0t = act + wave * 3 * (RPW * HIDN);
    __bf16* h1t = h0t + RPW * HIDN;
    __bf16* h2t = h1t + RPW * HIDN;
    {
        uint4 z; z.x = z.y = z.z = z.w = 0u;
        uint4* p = (uint4*)h0t;                 // 3 tiles contiguous = 768 uint4
        for (int i = lane; i < 768; i += 32) p[i] = z;
    }
    // initial position = obs_traj[-1]
    float* pw = posl + wave * 32;
    {
        int m = lane >> 1, j = lane & 1;
        pw[m * 2 + j] = obs_traj[((size_t)(OBSN - 1) * Bn + rb + m) * 2 + j];
    }
    __syncthreads();

    const __bf16* Whh0l = wreg;
    const __bf16* Wih1l = wreg + 1 * HIDN * HIDN;
    const __bf16* Whh1l = wreg + 2 * HIDN * HIDN;
    const __bf16* Wih2l = wreg + 3 * HIDN * HIDN;
    const __bf16* Whh2l = wreg + 4 * HIDN * HIDN;

    // ---- 12-step recurrence, fully on-chip, wave-private (no barriers) ----
    for (int step = 0; step < PREDN; ++step) {
        float px[8], py[8];
#pragma unroll
        for (int v = 0; v < 8; ++v) {
            int m = v + 8 * half;
            px[v] = pw[2 * m]; py[v] = pw[2 * m + 1];
        }

        v8f hn[8];
        // layer 0: h0 = tanh(ctxproj + pos@Wpos^T + h0@Whh0^T)
#pragma unroll
        for (int nt = 0; nt < 8; ++nt) {
            int n = (lane & 15) + nt * 16;
            v8f acc = cacc[nt];
            float w0 = wpos[2 * n], w1 = wpos[2 * n + 1];
#pragma unroll
            for (int v = 0; v < 8; ++v) acc[v] += px[v] * w0 + py[v] * w1;
            acc = gemm_tile(h0t, Whh0l, lane, nt, acc);
#pragma unroll
            for (int v = 0; v < 8; ++v) hn[nt][v] = fast_tanh(acc[v]);
        }
#pragma unroll
        for (int nt = 0; nt < 8; ++nt) storeD(h0t, lane, nt, hn[nt]);

        // layer 1: h1 = tanh(h0@Wih1^T + h1@Whh1^T + b1)
#pragma unroll
        for (int nt = 0; nt < 8; ++nt) {
            int n = (lane & 15) + nt * 16;
            float b_ = b1s[n];
            v8f acc;
#pragma unroll
            for (int v = 0; v < 8; ++v) acc[v] = b_;
            acc = gemm_tile(h0t, Wih1l, lane, nt, acc);
            acc = gemm_tile(h1t, Whh1l, lane, nt, acc);
#pragma unroll
            for (int v = 0; v < 8; ++v) hn[nt][v] = fast_tanh(acc[v]);
        }
#pragma unroll
        for (int nt = 0; nt < 8; ++nt) storeD(h1t, lane, nt, hn[nt]);

        // layer 2: h2 = tanh(h1@Wih2^T + h2@Whh2^T + b2)
#pragma unroll
        for (int nt = 0; nt < 8; ++nt) {
            int n = (lane & 15) + nt * 16;
            float b_ = b2s[n];
            v8f acc;
#pragma unroll
            for (int v = 0; v < 8; ++v) acc[v] = b_;
            acc = gemm_tile(h1t, Wih2l, lane, nt, acc);
            acc = gemm_tile(h2t, Whh2l, lane, nt, acc);
#pragma unroll
            for (int v = 0; v < 8; ++v) hn[nt][v] = fast_tanh(acc[v]);
        }
#pragma unroll
        for (int nt = 0; nt < 8; ++nt) storeD(h2t, lane, nt, hn[nt]);

        // fc head (K=2 output: VALU dot product; WMMA would waste a k=32 op)
        {
            int m = lane >> 1, j = lane & 1;
            float s = fcbl[j];
            const float* fr = fcwl + j * HIDN;
            const __bf16* hr = h2t + m * HIDN;
#pragma unroll 8
            for (int k = 0; k < HIDN; ++k) s += bf2f(hr[k]) * fr[k];
            out[((size_t)step * Bn + rb + m) * 2 + j] = s;
            pw[m * 2 + j] = s;   // feed-forward: next pos = detached output
        }
    }
}

extern "C" void kernel_launch(void* const* d_in, const int* in_sizes, int n_in,
                              void* d_out, int out_size, void* d_ws, size_t ws_size,
                              hipStream_t stream) {
    (void)n_in; (void)d_ws; (void)ws_size; (void)out_size;
    const float* raw_ctx  = (const float*)d_in[0];
    const float* obs_traj = (const float*)d_in[1];
    const float* Wih0 = (const float*)d_in[2];
    const float* Whh0 = (const float*)d_in[3];
    const float* bih0 = (const float*)d_in[4];
    const float* bhh0 = (const float*)d_in[5];
    const float* Wih1 = (const float*)d_in[6];
    const float* Whh1 = (const float*)d_in[7];
    const float* bih1 = (const float*)d_in[8];
    const float* bhh1 = (const float*)d_in[9];
    const float* Wih2 = (const float*)d_in[10];
    const float* Whh2 = (const float*)d_in[11];
    const float* bih2 = (const float*)d_in[12];
    const float* bhh2 = (const float*)d_in[13];
    const float* fcW  = (const float*)d_in[14];
    const float* fcb  = (const float*)d_in[15];
    float* out = (float*)d_out;

    const int Bn = in_sizes[0] / CTXN;              // 65536
    const int grid = (Bn + RPB - 1) / RPB;          // 512 workgroups
    // 160KB weights + 96KB activations + small params (ISA allows up to 320KB/WG)
    const size_t smem = 163840 + 98304 + (size_t)(256 + 128 + 128 + 256 + 4 + 256) * sizeof(float);

    rnn_head_kernel<<<grid, 256, smem, stream>>>(
        raw_ctx, obs_traj,
        Wih0, Whh0, bih0, bhh0,
        Wih1, Whh1, bih1, bhh1,
        Wih2, Whh2, bih2, bhh2,
        fcW, fcb, out, Bn);
}
